// MoELayer_73332271611934
// MI455X (gfx1250) — matrix operand order repaired and checked by reading, not compiled, not run on hardware
//
#include <hip/hip_runtime.h>
#include <hip/hip_bf16.h>
#include <math.h>

typedef __attribute__((ext_vector_type(16))) __bf16 bf16x16;
typedef __attribute__((ext_vector_type(8)))  __bf16 bf16x8;
typedef __attribute__((ext_vector_type(8)))  float  f32x8;

#define N_TOK 8192
#define C_DIM 1024
#define F_DIM 4096
#define E_NUM 8
#define BM 128
#define BN 128
#define BK 32
#define LDK 40   // padded LDS row stride in elements (80B -> conflict-free 16B lane loads)

// ---------------------------------------------------------------- utilities
__device__ __forceinline__ bf16x16 load_frag(const __bf16* p0, const __bf16* p1) {
    bf16x8 lo = *(const bf16x8*)p0;
    bf16x8 hi = *(const bf16x8*)p1;
    return __builtin_shufflevector(lo, hi, 0,1,2,3,4,5,6,7,8,9,10,11,12,13,14,15);
}

__device__ __forceinline__ float gelu_exact(float v) {
    return 0.5f * v * (1.0f + erff(v * 0.70710678118654752f));
}

// CDNA5 async global->LDS copy: per-lane 16B, tracked by ASYNCcnt (EXEC-masked)
__device__ __forceinline__ void async_copy16(unsigned lds_addr, const void* gptr) {
    unsigned long long ga = (unsigned long long)(uintptr_t)gptr;
    asm volatile("global_load_async_to_lds_b128 %0, %1, off"
                 :: "v"(lds_addr), "v"(ga) : "memory");
}
__device__ __forceinline__ void wait_async0() {
    asm volatile("s_wait_asynccnt 0x0" ::: "memory");
}
__device__ __forceinline__ unsigned lds_addr_of(const void* p) {
    return (unsigned)(uintptr_t)p;   // LDS offset is the low 32 bits of a shared-aperture flat address
}

// ---------------------------------------------------------------- kernel 0: zero counters
__global__ void zero_counts_kernel(int* counts) {
    if (threadIdx.x < E_NUM) counts[threadIdx.x] = 0;
}

// ---------------------------------------------------------------- prep A: x (f32) -> Xb (bf16)
__global__ __launch_bounds__(256)
void convert_x_kernel(const float* __restrict__ x, __bf16* __restrict__ Xb) {
    size_t base = ((size_t)blockIdx.x * 256 + threadIdx.x) * 8;
    float4 a = *(const float4*)(x + base);
    float4 b = *(const float4*)(x + base + 4);
    bf16x8 o;
    o[0] = (__bf16)a.x; o[1] = (__bf16)a.y; o[2] = (__bf16)a.z; o[3] = (__bf16)a.w;
    o[4] = (__bf16)b.x; o[5] = (__bf16)b.y; o[6] = (__bf16)b.z; o[7] = (__bf16)b.w;
    *(bf16x8*)(Xb + base) = o;
}

// ---------------------------------------------------------------- prep B: W [E][K][N] f32 -> Wt [E][N][K] bf16
// grid: (N/64, K/64, E), 256 threads
__global__ __launch_bounds__(256)
void convert_transpose_kernel(const float* __restrict__ src, __bf16* __restrict__ dst,
                              int K, int N) {
    const int e = blockIdx.z;
    src += (size_t)e * K * N;
    dst += (size_t)e * K * N;
    __shared__ __bf16 sT[64][72];
    const int t  = threadIdx.x;
    const int k0 = blockIdx.y * 64, n0 = blockIdx.x * 64;
    {   // coalesced load: 64 k-rows, 4 threads/row, 16 floats each
        int k = t >> 2, nq = (t & 3) * 16;
        const float* s = src + (size_t)(k0 + k) * N + n0 + nq;
#pragma unroll
        for (int i = 0; i < 4; ++i) {
            float4 v = ((const float4*)s)[i];
            sT[k][nq + 4*i + 0] = (__bf16)v.x;
            sT[k][nq + 4*i + 1] = (__bf16)v.y;
            sT[k][nq + 4*i + 2] = (__bf16)v.z;
            sT[k][nq + 4*i + 3] = (__bf16)v.w;
        }
    }
    __syncthreads();
    {   // coalesced store of transpose: 64 n-rows, 4 threads/row, 16 k each
        int n = t >> 2, kq = (t & 3) * 16;
        bf16x8 lo, hi;
#pragma unroll
        for (int i = 0; i < 8; ++i) { lo[i] = sT[kq + i][n]; hi[i] = sT[kq + 8 + i][n]; }
        __bf16* d = dst + (size_t)(n0 + n) * K + k0 + kq;
        *(bf16x8*)d       = lo;
        *(bf16x8*)(d + 8) = hi;
    }
}

// ---------------------------------------------------------------- kernel 1: gating + routing
__global__ __launch_bounds__(256)
void gate_kernel(const float* __restrict__ x, const float* __restrict__ gw,
                 const float* __restrict__ gb,
                 int* __restrict__ counts, int* __restrict__ slots,
                 float* __restrict__ slot_w) {
    int wid  = threadIdx.x >> 5;
    int lane = threadIdx.x & 31;
    int token = blockIdx.x * 8 + wid;
    if (token >= N_TOK) return;

    const float* xr = x + (size_t)token * C_DIM;
    float acc[E_NUM];
#pragma unroll
    for (int e = 0; e < E_NUM; ++e) acc[e] = 0.0f;

    for (int i = lane; i < C_DIM; i += 32) {
        float xv = xr[i];
        const float4* g = (const float4*)(gw + (size_t)i * E_NUM);
        float4 g0 = g[0], g1 = g[1];
        acc[0] += xv * g0.x; acc[1] += xv * g0.y;
        acc[2] += xv * g0.z; acc[3] += xv * g0.w;
        acc[4] += xv * g1.x; acc[5] += xv * g1.y;
        acc[6] += xv * g1.z; acc[7] += xv * g1.w;
    }
#pragma unroll
    for (int off = 16; off > 0; off >>= 1) {
#pragma unroll
        for (int e = 0; e < E_NUM; ++e)
            acc[e] += __shfl_xor(acc[e], off, 32);
    }

    if (lane == 0) {
        float l[E_NUM];
#pragma unroll
        for (int e = 0; e < E_NUM; ++e) l[e] = acc[e] + gb[e];
        int   e0 = 0; float l0 = l[0];
#pragma unroll
        for (int e = 1; e < E_NUM; ++e) if (l[e] > l0) { l0 = l[e]; e0 = e; }
        int   e1 = -1; float l1 = -3.4e38f;
#pragma unroll
        for (int e = 0; e < E_NUM; ++e) if (e != e0 && l[e] > l1) { l1 = l[e]; e1 = e; }
        float p1 = __expf(l1 - l0);
        float inv = 1.0f / (1.0f + p1);
        float w0 = inv, w1 = p1 * inv;

        int pos0 = atomicAdd(&counts[e0], 1);
        slots[e0 * N_TOK + pos0] = 2 * token;
        slot_w[2 * token] = w0;
        int pos1 = atomicAdd(&counts[e1], 1);
        slots[e1 * N_TOK + pos1] = 2 * token + 1;
        slot_w[2 * token + 1] = w1;
    }
}

// ---------------------------------------------------------------- GEMM core (bf16 A rows gathered, bf16 N-major B)
// grid: (NDIM/BN, N_TOK/BM, E), 256 threads = 8 waves (4 M x 2 N)
// double-buffered LDS filled by CDNA5 async global->LDS copies
template <int KDIM, bool IS_GEMM1>
__global__ __launch_bounds__(256)
void moe_gemm_kernel(const __bf16* __restrict__ Abase,   // Xb (gemm1) or H (gemm2)
                     const __bf16* __restrict__ Bt,      // Wt [E][NDIM][KDIM]
                     const float*  __restrict__ bias,    // [E][NDIM]
                     const int*    __restrict__ counts,
                     const int*    __restrict__ slots,
                     const float*  __restrict__ slot_w,  // used by gemm2
                     __bf16* __restrict__ Hout,          // gemm1 output
                     float*  __restrict__ Yout,          // gemm2 output
                     int NDIM) {
    const int e   = blockIdx.z;
    const int cnt = counts[e];
    const int m0  = blockIdx.y * BM;
    if (m0 >= cnt) return;
    const int n0  = blockIdx.x * BN;

    __shared__ __bf16 sA[2][BM * LDK];
    __shared__ __bf16 sB[2][BN * LDK];
    __shared__ int    sSlot[BM];
    __shared__ float  sW[BM];

    const int t = threadIdx.x;
    if (t < BM) {
        int r = m0 + t;
        int s = (r < cnt) ? slots[e * N_TOK + r] : -1;
        sSlot[t] = s;
        if (!IS_GEMM1) sW[t] = (s >= 0) ? slot_w[s] : 0.0f;
    }
    __syncthreads();

    const int wid = t >> 5, lane = t & 31;
    const int waveM = wid >> 1, waveN = wid & 1;
    const int h = lane >> 4, ln = lane & 15;

    f32x8 acc[2][4];
#pragma unroll
    for (int i = 0; i < 2; ++i)
#pragma unroll
        for (int j = 0; j < 4; ++j) acc[i][j] = (f32x8)0.0f;

    // staging maps: A: row t/2, 16 elems at (t&1)*16 ; B: n-row t/2, 16 k at (t&1)*16
    const int ar = t >> 1, ac = (t & 1) * 16;
    const int aslot = sSlot[ar];
    const __bf16* arow = (aslot >= 0)
        ? Abase + (size_t)(IS_GEMM1 ? (aslot >> 1) : aslot) * KDIM + ac
        : nullptr;
    const __bf16* brow = Bt + (size_t)e * NDIM * KDIM + (size_t)(n0 + ar) * KDIM + ac;

    const unsigned ldsA[2] = { lds_addr_of(sA[0] + ar * LDK + ac),
                               lds_addr_of(sA[1] + ar * LDK + ac) };
    const unsigned ldsB[2] = { lds_addr_of(sB[0] + ar * LDK + ac),
                               lds_addr_of(sB[1] + ar * LDK + ac) };

    // rows with no routed token: zero once in BOTH buffers (async copies never touch them)
    if (!arow) {
        const bf16x8 zv = (bf16x8)(__bf16)0.0f;
        *(bf16x8*)(sA[0] + ar * LDK + ac)     = zv;
        *(bf16x8*)(sA[0] + ar * LDK + ac + 8) = zv;
        *(bf16x8*)(sA[1] + ar * LDK + ac)     = zv;
        *(bf16x8*)(sA[1] + ar * LDK + ac + 8) = zv;
    }

    // ---- prologue: async-stage tile 0 into buffer 0
    if (arow) {
        async_copy16(ldsA[0],      arow);
        async_copy16(ldsA[0] + 16, arow + 8);
    }
    async_copy16(ldsB[0],      brow);
    async_copy16(ldsB[0] + 16, brow + 8);
    wait_async0();
    __syncthreads();

    int p = 0;
    for (int k0 = 0; k0 < KDIM; k0 += BK) {
        const int k1 = k0 + BK;
        const bool more = (k1 < KDIM);
        if (more) {  // async-stage next tile into the other buffer; overlaps with WMMAs below
            const int q = p ^ 1;
            if (arow) {
                async_copy16(ldsA[q],      arow + k1);
                async_copy16(ldsA[q] + 16, arow + k1 + 8);
            }
            async_copy16(ldsB[q],      brow + k1);
            async_copy16(ldsB[q] + 16, brow + k1 + 8);
        }

        bf16x16 afrag[2], bfrag[4];
#pragma unroll
        for (int tm = 0; tm < 2; ++tm) {
            const __bf16* base = sA[p] + (waveM * 32 + tm * 16 + ln) * LDK;
            afrag[tm] = load_frag(base + 8 * h, base + 16 + 8 * h);
        }
#pragma unroll
        for (int tn = 0; tn < 4; ++tn) {
            const __bf16* base = sB[p] + (waveN * 64 + tn * 16 + ln) * LDK;
            bfrag[tn] = load_frag(base + 8 * h, base + 16 + 8 * h);
        }
#pragma unroll
        for (int tm = 0; tm < 2; ++tm)
#pragma unroll
            for (int tn = 0; tn < 4; ++tn)
                acc[tm][tn] = __builtin_amdgcn_wmma_f32_16x16x32_bf16(
                    false, afrag[tm], false, bfrag[tn], (short)0, acc[tm][tn],
                    false, false);

        if (more) wait_async0();   // own async copies done; barrier orders across waves
        __syncthreads();
        p ^= 1;
    }

    // ---- epilogue
#pragma unroll
    for (int tn = 0; tn < 4; ++tn) {
        int gn = n0 + waveN * 64 + tn * 16 + ln;
        float bv = bias[e * NDIM + gn];
#pragma unroll
        for (int tm = 0; tm < 2; ++tm) {
#pragma unroll
            for (int v = 0; v < 8; ++v) {
                int r = waveM * 32 + tm * 16 + v + 8 * h;
                int slot = sSlot[r];
                if (slot >= 0) {
                    float val = acc[tm][tn][v] + bv;
                    if (IS_GEMM1) {
                        Hout[(size_t)slot * NDIM + gn] = (__bf16)gelu_exact(val);
                    } else {
                        Yout[(size_t)slot * NDIM + gn] = val * sW[r];
                    }
                }
            }
        }
    }
}

// ---------------------------------------------------------------- kernel 4: combine two expert slots per token
__global__ __launch_bounds__(256)
void combine_kernel(const float* __restrict__ Yb, float* __restrict__ out) {
    int i = blockIdx.x * blockDim.x + threadIdx.x;
    int base = i * 4;
    if (base >= N_TOK * C_DIM) return;
    int token = base / C_DIM;
    int col   = base % C_DIM;
    float4 a = *(const float4*)(Yb + (size_t)(2 * token)     * C_DIM + col);
    float4 b = *(const float4*)(Yb + (size_t)(2 * token + 1) * C_DIM + col);
    float4 o; o.x = a.x + b.x; o.y = a.y + b.y; o.z = a.z + b.z; o.w = a.w + b.w;
    *(float4*)(out + base) = o;
}

// ---------------------------------------------------------------- launch
extern "C" void kernel_launch(void* const* d_in, const int* in_sizes, int n_in,
                              void* d_out, int out_size, void* d_ws, size_t ws_size,
                              hipStream_t stream) {
    const float* x      = (const float*)d_in[0];
    const float* gate_w = (const float*)d_in[1];
    const float* gate_b = (const float*)d_in[2];
    const float* w1     = (const float*)d_in[3];
    const float* b1     = (const float*)d_in[4];
    const float* w2     = (const float*)d_in[5];
    const float* b2     = (const float*)d_in[6];
    float* out = (float*)d_out;

    char* ws = (char*)d_ws;
    size_t off = 0;
    int*    counts = (int*)(ws + off);   off += 256;
    int*    slots  = (int*)(ws + off);   off += (size_t)E_NUM * N_TOK * sizeof(int);
    float*  slot_w = (float*)(ws + off); off += (size_t)2 * N_TOK * sizeof(float);
    off = (off + 255) & ~(size_t)255;
    __bf16* Xb  = (__bf16*)(ws + off);   off += (size_t)N_TOK * C_DIM * sizeof(__bf16);
    __bf16* W1t = (__bf16*)(ws + off);   // [E][F][C] bf16, 64 MiB
    float*  Yb  = (float*)(ws + off);    // aliases W1t (dead after gemm1); 64 MiB
    off += (size_t)E_NUM * F_DIM * C_DIM * sizeof(__bf16);
    __bf16* W2t = (__bf16*)(ws + off);   off += (size_t)E_NUM * C_DIM * F_DIM * sizeof(__bf16);
    __bf16* H   = (__bf16*)(ws + off);   off += (size_t)2 * N_TOK * F_DIM * sizeof(__bf16);
    (void)ws_size; (void)in_sizes; (void)n_in; (void)out_size;

    zero_counts_kernel<<<1, 32, 0, stream>>>(counts);
    convert_x_kernel<<<(N_TOK * C_DIM) / (8 * 256), 256, 0, stream>>>(x, Xb);
    convert_transpose_kernel<<<dim3(F_DIM / 64, C_DIM / 64, E_NUM), 256, 0, stream>>>(
        w1, W1t, C_DIM, F_DIM);
    convert_transpose_kernel<<<dim3(C_DIM / 64, F_DIM / 64, E_NUM), 256, 0, stream>>>(
        w2, W2t, F_DIM, C_DIM);
    gate_kernel<<<N_TOK / 8, 256, 0, stream>>>(x, gate_w, gate_b, counts, slots, slot_w);

    moe_gemm_kernel<C_DIM, true><<<dim3(F_DIM / BN, N_TOK / BM, E_NUM), 256, 0, stream>>>(
        Xb, W1t, b1, counts, slots, slot_w, H, nullptr, F_DIM);
    moe_gemm_kernel<F_DIM, false><<<dim3(C_DIM / BN, N_TOK / BM, E_NUM), 256, 0, stream>>>(
        H, W2t, b2, counts, slots, slot_w, nullptr, Yb, C_DIM);

    combine_kernel<<<(N_TOK * C_DIM / 4 + 255) / 256, 256, 0, stream>>>(Yb, out);
}
